// RoPEMHSA_8796093022277
// MI455X (gfx1250) — matrix operand order repaired and checked by reading, not compiled, not run on hardware
//
#include <hip/hip_runtime.h>
#include <hip/hip_bf16.h>
#include <stdint.h>

// ---------------------------------------------------------------------------
// RoPE MHSA for gfx1250 (MI455X).  B=4, S=2048, D=512, H=8, HD=64.
// Matmuls via v_wmma_f32_16x16x32_bf16 (wave32 WMMA).
// GEMM A-tiles staged by the Tensor Data Mover (tensor_load_to_lds, TENSORcnt),
// B-tiles by global_load_async_to_lds_b128 (ASYNCcnt). Flash-style softmax.
// ---------------------------------------------------------------------------

#define BATCH 4
#define SEQ   2048
#define DIM   512
#define NH    8
#define HD    64
#define BH    (BATCH * NH)      // 32
#define ROWS  (BATCH * SEQ)     // 8192

typedef __attribute__((ext_vector_type(16))) __bf16 bf16x16;
typedef __attribute__((ext_vector_type(8)))  __bf16 bf16x8;
typedef __attribute__((ext_vector_type(8)))  float  f32x8;
typedef __attribute__((ext_vector_type(4)))  uint32_t u32x4;
typedef __attribute__((ext_vector_type(8)))  uint32_t u32x8;

__device__ __forceinline__ bf16x16 cat8(bf16x8 lo, bf16x8 hi) {
  return __builtin_shufflevector(lo, hi, 0,1,2,3,4,5,6,7,8,9,10,11,12,13,14,15);
}

// A-fragment (16x32, 16-bit): row = lane&15, K in {k0..k0+7, k0+16..k0+23},
// k0 = 8*(lane>=16).
__device__ __forceinline__ bf16x16 load_frag_a(const __bf16* base, int ld, int lane) {
  int row = lane & 15;
  int k0  = (lane >> 4) << 3;          // 0 or 8
  const __bf16* p = base + (size_t)row * ld;
  bf16x8 lo = *(const bf16x8*)(p + k0);
  bf16x8 hi = *(const bf16x8*)(p + k0 + 16);
  return cat8(lo, hi);
}

// B-fragment (32x16, 16-bit) from K-contiguous storage Bt[n][k]:
// N=lane&15, K in {k0..k0+7, k0+16..k0+23}, k0 = 8*(lane>=16).
__device__ __forceinline__ bf16x16 load_frag_b(const __bf16* base, int ld, int lane) {
  int n  = lane & 15;
  int k0 = (lane >> 4) << 3;           // 0 or 8
  const __bf16* p = base + (size_t)n * ld + k0;
  bf16x8 lo = *(const bf16x8*)(p);
  bf16x8 hi = *(const bf16x8*)(p + 16);
  return cat8(lo, hi);
}

__device__ __forceinline__ f32x8 wmma_bf16(bf16x16 a, bf16x16 b, f32x8 c) {
  return __builtin_amdgcn_wmma_f32_16x16x32_bf16(false, a, false, b, (short)0, c,
                                                 false, false);
}

// Async memory->LDS copy of 16 bytes per lane (gfx1250, tracked by ASYNCcnt).
__device__ __forceinline__ void async_copy_b128(uint32_t ldsOff, const void* gptr) {
  asm volatile("global_load_async_to_lds_b128 %0, %1, off"
               :: "v"(ldsOff), "v"((uint64_t)(uintptr_t)gptr)
               : "memory");
}
__device__ __forceinline__ void async_wait0() {
  asm volatile("s_wait_asynccnt 0" ::: "memory");
}

// ---------------------------------------------------------------------------
// Tensor Data Mover: 2-D tile load Global->LDS (bf16 elements).
//   tileW x tileH elements, source row stride = rowStride elements,
//   LDS padded by padAmt DWORDs every padIntv DWORDs (encoded per D# spec).
// One wave issues this; completion tracked by TENSORcnt.
// ---------------------------------------------------------------------------
__device__ __forceinline__ void tdm_load_2d_bf16(uint32_t ldsOff, const void* gptr,
                                                 uint32_t tileW, uint32_t tileH,
                                                 uint32_t rowStride) {
  union { uint32_t d[4]; u32x4 v; } g0;
  union { uint32_t d[8]; u32x8 v; } g1;
  uint64_t ga = (uint64_t)(uintptr_t)gptr;
  // group0: count=1 | lds_addr | global_addr[56:0] | type=2
  g0.d[0] = 1u;
  g0.d[1] = ldsOff;
  g0.d[2] = (uint32_t)ga;
  g0.d[3] = (uint32_t)(ga >> 32) | (2u << 30);
  // group1
  const uint32_t dim0 = rowStride;       // tensor width (elements)
  const uint32_t dim1 = 0x100000u;       // generous height (no OOB clamp)
  // data_size=1 (2B), pad_enable, pad_interval=3 (16 DWORDs = one 64B row),
  // pad_amount=3 (4 DWORDs = 16B)  ->  LDS row stride = 80B = LDT bf16
  g1.d[0] = (1u << 16) | (1u << 20) | (3u << 22) | (3u << 25);
  g1.d[1] = (dim0 & 0xFFFFu) << 16;                      // dim0 lo16 @ b63:48
  g1.d[2] = (dim0 >> 16) | ((dim1 & 0xFFFFu) << 16);     // dim0 hi / dim1 lo
  g1.d[3] = (dim1 >> 16) | (tileW << 16);                // dim1 hi / tile_dim0
  g1.d[4] = tileH;                                       // tile_dim1 (dim2=0)
  g1.d[5] = rowStride;                                   // tensor_dim0_stride lo
  g1.d[6] = 0;                                           // stride hi / dim1_stride
  g1.d[7] = 0;
  asm volatile("tensor_load_to_lds %0, %1"
               :: "s"(g0.v), "s"(g1.v)
               : "memory");
}

// ---------------------------------------------------------------------------
// elementwise converts
// ---------------------------------------------------------------------------
__global__ void f32_to_bf16_kernel(const float* __restrict__ in,
                                   __bf16* __restrict__ out, int n) {
  int i = blockIdx.x * blockDim.x + threadIdx.x;
  if (i < n) out[i] = (__bf16)in[i];
}

// weight [K][N] f32 -> wt [N][K] bf16  (K = N = 512)
__global__ void wtrans_bf16_kernel(const float* __restrict__ w,
                                   __bf16* __restrict__ wt) {
  int i = blockIdx.x * blockDim.x + threadIdx.x;   // 512*512
  if (i >= DIM * DIM) return;
  int k = i >> 9;
  int n = i & 511;
  wt[(size_t)n * DIM + k] = (__bf16)w[i];
}

// ---------------------------------------------------------------------------
// RoPE:  Qf32[B,S,D] -> bf16 head-major [BH][S][HD]
// ---------------------------------------------------------------------------
__global__ void rope_kernel(const float* __restrict__ X,
                            const float* __restrict__ cosE,
                            const float* __restrict__ sinE,
                            __bf16* __restrict__ Outbf) {
  int i = blockIdx.x * blockDim.x + threadIdx.x;   // ROWS*DIM
  if (i >= ROWS * DIM) return;
  int col = i & (DIM - 1);
  int row = i >> 9;            // b*S + s
  int s   = row & (SEQ - 1);
  int b   = row >> 11;
  float x  = X[i];
  float xr = (col < DIM / 2) ? -X[i + DIM / 2] : X[i - DIM / 2];
  float c  = cosE[(size_t)s * DIM + col];
  float sn = sinE[(size_t)s * DIM + col];
  float v  = x * c + xr * sn;
  int h  = col >> 6;
  int hd = col & 63;
  Outbf[(((size_t)(b * NH + h) * SEQ + s) << 6) + hd] = (__bf16)v;
}

// ---------------------------------------------------------------------------
// Tiled bf16 WMMA GEMM:  C[M,N] = A[M,K] * Bt[N,K]^T, f32 accumulate.
// A tile: one TDM tensor_load_to_lds per k-step (wave 0, TENSORcnt).
// B tile: per-lane global_load_async_to_lds_b128 (ASYNCcnt).
// mode 0: store f32 row-major ; mode 1: store bf16 to Vt[(b*8+h)*64+hd][s].
// ---------------------------------------------------------------------------
#define BM 128
#define BN 128
#define BK 32
#define LDT 40   // padded LDS row stride (bf16 elements) == TDM pad config

__global__ __launch_bounds__(256) void gemm_bf16_kernel(
    const __bf16* __restrict__ A, const __bf16* __restrict__ Bt,
    void* __restrict__ Cout, int M, int N, int K, int mode) {
  __shared__ __bf16 sA[BM * LDT];
  __shared__ __bf16 sB[BN * LDT];
  int tid  = threadIdx.x;
  int lane = tid & 31;
  int wave = tid >> 5;
  int mb = blockIdx.x * BM;
  int nb = blockIdx.y * BN;
  int wr = wave & 3;   // 4 row groups of 32
  int wc = wave >> 2;  // 2 col groups of 64

  f32x8 acc[2][4] = {};

  for (int kb = 0; kb < K; kb += BK) {
    // A tile (128x32) via Tensor Data Mover: single wave-level DMA
    if (wave == 0) {
      tdm_load_2d_bf16((uint32_t)(uintptr_t)&sA[0],
                       &A[(size_t)mb * K + kb], BK, BM, K);
    }
    // B tile (128x32): per-lane async 16B copies
    #pragma unroll
    for (int it = 0; it < 2; ++it) {
      int c   = tid + it * 256;          // chunk id, 512 chunks of 8 bf16
      int row = c >> 2;
      int ch  = (c & 3) << 3;
      async_copy_b128((uint32_t)(uintptr_t)&sB[row * LDT + ch],
                      &Bt[(size_t)(nb + row) * K + kb + ch]);
    }
    if (kb + BK < K) {
      __builtin_prefetch(&Bt[(size_t)(nb + (tid >> 2)) * K + kb + BK], 0, 1);
    }
    async_wait0();                          // own async copies landed
    if (wave == 0) __builtin_amdgcn_s_wait_tensorcnt(0);  // TDM landed
    __syncthreads();                        // everyone's staging visible

    bf16x16 af[2], bfr[4];
    #pragma unroll
    for (int mt = 0; mt < 2; ++mt)
      af[mt] = load_frag_a(&sA[(wr * 32 + mt * 16) * LDT], LDT, lane);
    #pragma unroll
    for (int nt = 0; nt < 4; ++nt)
      bfr[nt] = load_frag_b(&sB[(wc * 64 + nt * 16) * LDT], LDT, lane);
    #pragma unroll
    for (int mt = 0; mt < 2; ++mt)
      #pragma unroll
      for (int nt = 0; nt < 4; ++nt)
        acc[mt][nt] = wmma_bf16(af[mt], bfr[nt], acc[mt][nt]);
    __syncthreads();
  }

  int rbase = (lane >> 4) << 3;
  int col0  = lane & 15;
  #pragma unroll
  for (int mt = 0; mt < 2; ++mt) {
    #pragma unroll
    for (int nt = 0; nt < 4; ++nt) {
      #pragma unroll
      for (int r = 0; r < 8; ++r) {
        int grow = mb + wr * 32 + mt * 16 + rbase + r;
        int gcol = nb + wc * 64 + nt * 16 + col0;
        float v  = acc[mt][nt][r];
        if (mode == 0) {
          ((float*)Cout)[(size_t)grow * N + gcol] = v;
        } else {
          int b = grow >> 11, s = grow & 2047;
          int h = gcol >> 6,  hd = gcol & 63;
          ((__bf16*)Cout)[(((size_t)((b << 3) + h) << 6) + hd) * SEQ + s] =
              (__bf16)v;
        }
      }
    }
  }
}

// ---------------------------------------------------------------------------
// Flash attention: Q,K bf16 [BH][S][HD], Vt bf16 [BH][HD][S], mask int [B][S][S].
// Block = 128 threads (4 waves), each wave owns 16 query rows; 64-key tiles.
// Out: bf16 [B][S][D] (head-interleaved), ready for the output-proj GEMM.
// ---------------------------------------------------------------------------
#define PLD 80   // padded LDS stride for the P tile

__global__ __launch_bounds__(128) void attn_kernel(
    const __bf16* __restrict__ Q, const __bf16* __restrict__ Kb,
    const __bf16* __restrict__ Vt, const int* __restrict__ mask,
    __bf16* __restrict__ Out) {
  __shared__ __bf16 sP[4][16 * PLD];
  int tid  = threadIdx.x;
  int lane = tid & 31;
  int wave = tid >> 5;
  int bh = blockIdx.x;
  int b  = bh >> 3;
  int h  = bh & 7;
  int qb = blockIdx.y * 64 + wave * 16;    // query-row base for this wave
  int rbase = (lane >> 4) << 3;
  int col0  = lane & 15;

  const __bf16* Qbase = Q + ((size_t)bh * SEQ + qb) * HD;
  bf16x16 aq[2];
  aq[0] = load_frag_a(Qbase, HD, lane);
  aq[1] = load_frag_a(Qbase + 32, HD, lane);

  f32x8 O[4] = {};
  float mrow[8], lrow[8], arow[8];
  #pragma unroll
  for (int r = 0; r < 8; ++r) { mrow[r] = -__builtin_inff(); lrow[r] = 0.f; }

  const float scale = 0.125f;  // 1/sqrt(HD)

  for (int kb = 0; kb < SEQ; kb += 64) {
    // ---- scores S = Q * K^T (16 x 64) ----
    f32x8 sc[4];
    #pragma unroll
    for (int nt = 0; nt < 4; ++nt) {
      f32x8 a = {};
      #pragma unroll
      for (int c = 0; c < 2; ++c) {
        bf16x16 bk = load_frag_b(
            Kb + ((size_t)bh * SEQ + kb + nt * 16) * HD + c * 32, HD, lane);
        a = wmma_bf16(aq[c], bk, a);
      }
      sc[nt] = a;
    }

    // ---- mask + online softmax stats ----
    #pragma unroll
    for (int r = 0; r < 8; ++r) {
      int q = qb + rbase + r;
      const int* mp = mask + ((size_t)b * SEQ + q) * SEQ + kb;
      float rmax = -__builtin_inff();
      #pragma unroll
      for (int nt = 0; nt < 4; ++nt) {
        int k   = nt * 16 + col0;
        float s = (mp[k] != 0) ? sc[nt][r] * scale : -__builtin_inff();
        sc[nt][r] = s;
        rmax = fmaxf(rmax, s);
      }
      #pragma unroll
      for (int off = 1; off < 16; off <<= 1)
        rmax = fmaxf(rmax, __shfl_xor(rmax, off, 32));
      float mnew  = fmaxf(mrow[r], rmax);
      float alpha = (mnew == mrow[r]) ? 1.0f : __expf(mrow[r] - mnew);
      float mden  = fmaxf(mnew, -3.0e38f);  // keeps exp(-inf - mden) == 0
      float psum = 0.f;
      #pragma unroll
      for (int nt = 0; nt < 4; ++nt) {
        float p = __expf(sc[nt][r] - mden);  // masked entries -> exactly 0
        psum += p;
        sP[wave][(rbase + r) * PLD + nt * 16 + col0] = (__bf16)p;
      }
      #pragma unroll
      for (int off = 1; off < 16; off <<= 1) psum += __shfl_xor(psum, off, 32);
      lrow[r] = lrow[r] * alpha + psum;
      mrow[r] = mnew;
      arow[r] = alpha;
    }

    // ---- rescale O accumulators ----
    #pragma unroll
    for (int r = 0; r < 8; ++r) {
      float a = arow[r];
      #pragma unroll
      for (int nt2 = 0; nt2 < 4; ++nt2) O[nt2][r] *= a;
    }

    // ---- O += P * V ----
    #pragma unroll
    for (int kc = 0; kc < 2; ++kc) {
      bf16x16 ap = load_frag_a(&sP[wave][kc * 32], PLD, lane);
      #pragma unroll
      for (int nt2 = 0; nt2 < 4; ++nt2) {
        bf16x16 bv = load_frag_b(
            Vt + ((size_t)bh * HD + nt2 * 16) * SEQ + kb + kc * 32, SEQ, lane);
        O[nt2] = wmma_bf16(ap, bv, O[nt2]);
      }
    }
  }

  // ---- normalize + store head-interleaved bf16 ----
  #pragma unroll
  for (int r = 0; r < 8; ++r) {
    float inv = (lrow[r] > 0.f) ? 1.0f / lrow[r] : 0.f;
    int q = qb + rbase + r;
    #pragma unroll
    for (int nt2 = 0; nt2 < 4; ++nt2) {
      int hd = nt2 * 16 + col0;
      Out[((size_t)(b * SEQ + q)) * DIM + h * HD + hd] = (__bf16)(O[nt2][r] * inv);
    }
  }
}

// ---------------------------------------------------------------------------
// host-side orchestration
// ---------------------------------------------------------------------------
extern "C" void kernel_launch(void* const* d_in, const int* in_sizes, int n_in,
                              void* d_out, int out_size, void* d_ws, size_t ws_size,
                              hipStream_t stream) {
  (void)in_sizes; (void)n_in; (void)out_size; (void)ws_size;
  const float* x    = (const float*)d_in[0];
  const float* cosE = (const float*)d_in[1];
  const float* sinE = (const float*)d_in[2];
  const int*   mask = (const int*)d_in[3];
  const float* wq   = (const float*)d_in[4];
  const float* wk   = (const float*)d_in[5];
  const float* wv   = (const float*)d_in[6];
  const float* wo   = (const float*)d_in[7];
  float* out = (float*)d_out;

  char* ws = (char*)d_ws;
  const size_t SZ_XBF = (size_t)ROWS * DIM * 2;       // 8 MB
  const size_t SZ_W   = (size_t)DIM * DIM * 2;        // 0.5 MB each
  const size_t SZ_QF  = (size_t)ROWS * DIM * 4;       // 16 MB each
  const size_t SZ_QBF = (size_t)ROWS * DIM * 2;       // 8 MB each
  size_t off = 0;
  __bf16* xbf = (__bf16*)(ws + off); off += SZ_XBF;
  __bf16* wqT = (__bf16*)(ws + off); off += SZ_W;
  __bf16* wkT = (__bf16*)(ws + off); off += SZ_W;
  __bf16* wvT = (__bf16*)(ws + off); off += SZ_W;
  __bf16* woT = (__bf16*)(ws + off); off += SZ_W;
  float*  Qf  = (float*)(ws + off);  size_t qf_off = off; off += SZ_QF;
  float*  Kf  = (float*)(ws + off);  off += SZ_QF;
  __bf16* Qbf = (__bf16*)(ws + off); off += SZ_QBF;
  __bf16* Kbf = (__bf16*)(ws + off); off += SZ_QBF;
  __bf16* Vt  = (__bf16*)(ws + off); off += SZ_QBF;
  __bf16* attn = (__bf16*)(ws + qf_off);   // alias Qf (free after rope)

  // 1) converts
  {
    int n = ROWS * DIM;
    f32_to_bf16_kernel<<<(n + 255) / 256, 256, 0, stream>>>(x, xbf, n);
    int nw = DIM * DIM;
    wtrans_bf16_kernel<<<(nw + 255) / 256, 256, 0, stream>>>(wq, wqT);
    wtrans_bf16_kernel<<<(nw + 255) / 256, 256, 0, stream>>>(wk, wkT);
    wtrans_bf16_kernel<<<(nw + 255) / 256, 256, 0, stream>>>(wv, wvT);
    wtrans_bf16_kernel<<<(nw + 255) / 256, 256, 0, stream>>>(wo, woT);
  }

  // 2) projections
  dim3 ggrid(ROWS / BM, DIM / BN);
  gemm_bf16_kernel<<<ggrid, 256, 0, stream>>>(xbf, wqT, Qf, ROWS, DIM, DIM, 0);
  gemm_bf16_kernel<<<ggrid, 256, 0, stream>>>(xbf, wkT, Kf, ROWS, DIM, DIM, 0);
  gemm_bf16_kernel<<<ggrid, 256, 0, stream>>>(xbf, wvT, Vt, ROWS, DIM, DIM, 1);

  // 3) RoPE -> head-major bf16
  {
    int n = ROWS * DIM;
    rope_kernel<<<(n + 255) / 256, 256, 0, stream>>>(Qf, cosE, sinE, Qbf);
    rope_kernel<<<(n + 255) / 256, 256, 0, stream>>>(Kf, cosE, sinE, Kbf);
  }

  // 4) flash attention
  attn_kernel<<<dim3(BH, SEQ / 64), 128, 0, stream>>>(Qbf, Kbf, Vt, mask, attn);

  // 5) output projection (f32 out)
  gemm_bf16_kernel<<<ggrid, 256, 0, stream>>>(attn, woT, out, ROWS, DIM, DIM, 0);
}
